// GNNEncoder_15496242004308
// MI455X (gfx1250) — matrix operand order repaired and checked by reading, not compiled, not run on hardware
//
#include <hip/hip_runtime.h>

#define B_   32
#define N_   19
#define C_   512
#define EPS_ 1e-5f

typedef __attribute__((ext_vector_type(16))) __bf16 v16bf;
typedef __attribute__((ext_vector_type(8)))  float  v8f;

// ---------------------------------------------------------------------------
// Fragment loaders (f32 memory -> bf16 WMMA fragments, contiguous k = 16*half + s in A and B)
// ---------------------------------------------------------------------------
__device__ __forceinline__ v16bf load_frag_a(const float* __restrict__ p, int half)
{
  v16bf a;
#pragma unroll
  for (int i = 0; i < 8; ++i) {
    // A 16x32: lanes 0-15 K=0..15, lanes 16-31 K=16..31, 2 K-values per VGPR
    const int ka = 16 * half + 2 * i;
    const float2 f = *(const float2*)(p + ka);
    a[2 * i]     = (__bf16)f.x;
    a[2 * i + 1] = (__bf16)f.y;
  }
  return a;
}

__device__ __forceinline__ v16bf load_frag_b(const float* __restrict__ p, int half)
{
  v16bf b;
#pragma unroll
  for (int i = 0; i < 8; ++i) {
    // B 32x16: lanes 0-15 K=0..15, lanes 16-31 K=16..31, 2 K-values per VGPR
    const int kb = 16 * half + 2 * i;
    const float2 f = *(const float2*)(p + kb);
    b[2 * i]     = (__bf16)f.x;
    b[2 * i + 1] = (__bf16)f.y;
  }
  return b;
}

// ---------------------------------------------------------------------------
// WMMA GEMM:  C[M,Nout] = A[M,K] @ W[Nout,K]^T  (+bias) (+BN eval) (+ReLU)
// One wave per 32x32 output block: 2 A-frags x 2 B-frags -> 4 WMMAs / K-step.
// K multiple of 32, M and Nout multiples of 32.
// flags: bit0 = ReLU, bit1 = BN (scale = g*rsqrt(v+eps), shift = b - m*scale)
// ---------------------------------------------------------------------------
__global__ __launch_bounds__(32) void wmma_gemm_bf16(
    const float* __restrict__ A, int lda,
    const float* __restrict__ W, int ldb,
    float* __restrict__ C, int ldc, int K,
    const float* __restrict__ bias,
    const float* __restrict__ bn_m, const float* __restrict__ bn_v,
    const float* __restrict__ bn_g, const float* __restrict__ bn_b,
    int flags)
{
  const int lane = threadIdx.x;
  const int half = lane >> 4;
  const int l16  = lane & 15;

  const float* Ap0 = A + (long)(blockIdx.y * 32 + l16) * lda;
  const float* Ap1 = Ap0 + (long)16 * lda;
  const float* Wp0 = W + (long)(blockIdx.x * 32 + l16) * ldb;
  const float* Wp1 = Wp0 + (long)16 * ldb;

  v8f acc00 = {}, acc01 = {}, acc10 = {}, acc11 = {};
  for (int k0 = 0; k0 < K; k0 += 32) {
    const v16bf a0 = load_frag_a(Ap0 + k0, half);
    const v16bf a1 = load_frag_a(Ap1 + k0, half);
    const v16bf b0 = load_frag_b(Wp0 + k0, half);
    const v16bf b1 = load_frag_b(Wp1 + k0, half);
    acc00 = __builtin_amdgcn_wmma_f32_16x16x32_bf16(false, a0, false, b0, (short)0, acc00, false, false);
    acc01 = __builtin_amdgcn_wmma_f32_16x16x32_bf16(false, a0, false, b1, (short)0, acc01, false, false);
    acc10 = __builtin_amdgcn_wmma_f32_16x16x32_bf16(false, a1, false, b0, (short)0, acc10, false, false);
    acc11 = __builtin_amdgcn_wmma_f32_16x16x32_bf16(false, a1, false, b1, (short)0, acc11, false, false);
  }

  // per-column epilogue parameters for the two 16-col tiles
  const int col0 = blockIdx.x * 32 + l16;
  const int col1 = col0 + 16;
  float bv0 = bias ? bias[col0] : 0.0f;
  float bv1 = bias ? bias[col1] : 0.0f;
  float sc0 = 1.0f, sh0 = 0.0f, sc1 = 1.0f, sh1 = 0.0f;
  if (flags & 2) {
    const float rs0 = rsqrtf(bn_v[col0] + EPS_);
    sc0 = bn_g[col0] * rs0;  sh0 = bn_b[col0] - bn_m[col0] * sc0;
    const float rs1 = rsqrtf(bn_v[col1] + EPS_);
    sc1 = bn_g[col1] * rs1;  sh1 = bn_b[col1] - bn_m[col1] * sc1;
  }

  const int row0 = blockIdx.y * 32 + half * 8;   // C/D: VGPR r -> row half*8+r
#pragma unroll
  for (int r = 0; r < 8; ++r) {
    float y;
    y = (acc00[r] + bv0) * sc0 + sh0; if (flags & 1) y = fmaxf(y, 0.0f);
    C[(long)(row0 + r) * ldc + col0] = y;
    y = (acc01[r] + bv1) * sc1 + sh1; if (flags & 1) y = fmaxf(y, 0.0f);
    C[(long)(row0 + r) * ldc + col1] = y;
    y = (acc10[r] + bv0) * sc0 + sh0; if (flags & 1) y = fmaxf(y, 0.0f);
    C[(long)(row0 + 16 + r) * ldc + col0] = y;
    y = (acc11[r] + bv1) * sc1 + sh1; if (flags & 1) y = fmaxf(y, 0.0f);
    C[(long)(row0 + 16 + r) * ldc + col1] = y;
  }
}

// ---- xt[b, c*N + n] = x[b, n, c]  (for conv1 einsum with contiguous K) ----
__global__ void k_transpose_x(const float* __restrict__ x, float* __restrict__ xt)
{
  const int idx = blockIdx.x * 256 + threadIdx.x;   // < B*N*C
  const int b = idx / (N_ * C_);
  const int r = idx - b * (N_ * C_);
  const int n = r / C_;
  const int c = r - n * C_;
  xt[(long)b * (C_ * N_) + c * N_ + n] = x[idx];
}

// ---- out[b,:] = LN(A[b,:] + Bv[b,:]) * g + beta ; one block per batch row ----
__global__ void k_add_ln(const float* __restrict__ A, const float* __restrict__ Bv,
                         const float* __restrict__ g, const float* __restrict__ beta,
                         float* __restrict__ out)
{
  __shared__ float sh[256];
  const int b = blockIdx.x, t = threadIdx.x;
  const int base = b * C_;
  const float z0 = A[base + t] + Bv[base + t];
  const float z1 = A[base + 256 + t] + Bv[base + 256 + t];
  sh[t] = z0 + z1;
  __syncthreads();
  for (int o = 128; o > 0; o >>= 1) { if (t < o) sh[t] += sh[t + o]; __syncthreads(); }
  const float mu = sh[0] * (1.0f / C_);
  __syncthreads();
  const float d0 = z0 - mu, d1 = z1 - mu;
  sh[t] = d0 * d0 + d1 * d1;
  __syncthreads();
  for (int o = 128; o > 0; o >>= 1) { if (t < o) sh[t] += sh[t + o]; __syncthreads(); }
  const float inv = rsqrtf(sh[0] * (1.0f / C_) + EPS_);
  out[base + t]       = g[t] * d0 * inv + beta[t];
  out[base + 256 + t] = g[256 + t] * d1 * inv + beta[256 + t];
}

// ---- PNA stats of P2 over source nodes u: mean / min / max / std ----
__global__ void k_pna_stats(const float* __restrict__ P2, float* __restrict__ sm,
                            float* __restrict__ smin, float* __restrict__ smax,
                            float* __restrict__ sstd)
{
  const int idx = blockIdx.x * 256 + threadIdx.x;   // b*C + c
  const int b = idx >> 9, c = idx & (C_ - 1);
  const float* p = P2 + (long)b * N_ * C_ + c;
  float mn = 3.4e38f, mx = -3.4e38f, s = 0.f, sq = 0.f;
#pragma unroll
  for (int u = 0; u < N_; ++u) {
    const float v = p[(long)u * C_];
    s += v; sq += v * v; mn = fminf(mn, v); mx = fmaxf(mx, v);
  }
  const float mean = s * (1.0f / N_);
  const float var  = sq * (1.0f / N_) - mean * mean;
  sm[idx] = mean; smin[idx] = mn; smax[idx] = mx;
  sstd[idx] = sqrtf(fmaxf(var, 0.f) + EPS_);
}

// ---- agg[row, 0:5C] = [x, P1+P3+mean, P1+P3+min, P1+P3+max, std] ----
__global__ void k_assemble(const float* __restrict__ xin, const float* __restrict__ P1,
                           const float* __restrict__ P3, const float* __restrict__ sm,
                           const float* __restrict__ smin, const float* __restrict__ smax,
                           const float* __restrict__ sstd, float* __restrict__ agg)
{
  const long idx = (long)blockIdx.x * 256 + threadIdx.x;  // < 608*2560
  const int row = (int)(idx / (5 * C_));                  // b*N + v
  const int j   = (int)(idx - (long)row * (5 * C_));
  const int b   = row / N_;
  const int seg = j >> 9, c = j & (C_ - 1);
  float val;
  if (seg == 0)      val = xin[(long)row * C_ + c];
  else if (seg == 4) val = sstd[b * C_ + c];
  else {
    const float base = P1[(long)row * C_ + c] + P3[b * C_ + c];
    const float* sv = (seg == 1) ? sm : (seg == 2) ? smin : smax;
    val = base + sv[b * C_ + c];
  }
  agg[idx] = val;
}

// ---- GraphNorm over nodes + ReLU + residual(s) ----
__global__ void k_graphnorm(const float* __restrict__ O, const float* __restrict__ xin,
                            const float* __restrict__ extra, const float* __restrict__ ga,
                            const float* __restrict__ gb, const float* __restrict__ gg,
                            float* __restrict__ out)
{
  const int idx = blockIdx.x * 256 + threadIdx.x;   // b*C + c
  const int b = idx >> 9, c = idx & (C_ - 1);
  const float* p = O + (long)b * N_ * C_ + c;
  float s = 0.f;
#pragma unroll
  for (int v = 0; v < N_; ++v) s += p[(long)v * C_];
  const float mu = s * (1.0f / N_) * ga[c];         // gn_a * mean
  float vs = 0.f;
#pragma unroll
  for (int v = 0; v < N_; ++v) { const float d = p[(long)v * C_] - mu; vs += d * d; }
  const float inv = rsqrtf(vs * (1.0f / N_) + EPS_);
  const float gc = gg[c], bc = gb[c];
  for (int v = 0; v < N_; ++v) {
    const long o = (long)(b * N_ + v) * C_ + c;
    const float d = p[(long)v * C_] - mu;
    float y = fmaxf(gc * d * inv + bc, 0.f) + xin[o];
    if (extra) y += extra[o];
    out[o] = y;
  }
}

// ---------------------------------------------------------------------------
// Host side
// ---------------------------------------------------------------------------
struct GemP  { const float *arm_in_b,*arm_in_w,*arm_out_b,*arm_out_w,
                           *fam_in_b,*fam_in_w,*fam_out_b,*fam_out_w,
                           *ff1_b,*ff1_w,*ff2_b,*ff2_w,*ln_b,*ln_g; };
struct GfemP { const float *bn1_b,*bn1_g,*bn1_m,*bn1_v,*bn2_b,*bn2_g,*bn2_m,*bn2_v,
                           *conv1_b,*conv1_w,*conv2_b,*conv2_w; };
struct GnnP  { const float *edge_b,*edge_w,*gn_a,*gn_b,*gn_g,
                           *lin_b,*lin_w,*post_b,*post_w,*pre_b,*pre_w; };

static inline void gemm(const float* A, int lda, const float* W, int ldb,
                        float* C, int ldc, int M, int Nout, int K,
                        const float* bias, const float* m, const float* v,
                        const float* g, const float* b, int flags, hipStream_t s)
{
  dim3 grid(Nout / 32, M / 32);
  wmma_gemm_bf16<<<grid, 32, 0, s>>>(A, lda, W, ldb, C, ldc, K, bias, m, v, g, b, flags);
}

// workspace offsets (floats)
enum : long {
  XT = 0,           GF1 = 311296,  GLOB = 376832, T0 = 393216, T1 = 409600,
  T2 = 425984,      H1 = 442368,   EV = 507904,   P1O = 524288, P2O = 835584,
  RM = 1146880, RMIN = 1163264, RMAX = 1179648, RSTD = 1196032,
  AGG = 1212416,    O1 = 2768896,  O2 = 3080192,  FV1 = 3391488
};

static void run_stage(const float* xin, const GfemP& g, const GemP& e, const GnnP& n,
                      const float* extra_res, float* outbuf, float* ws, hipStream_t s)
{
  // ---- GFEM: global feature [B, C] ----
  k_transpose_x<<<(B_ * N_ * C_) / 256, 256, 0, s>>>(xin, ws + XT);
  gemm(ws + XT, C_ * N_, g.conv1_w, C_ * N_, ws + GF1, 4 * C_, B_, 4 * C_, C_ * N_,
       g.conv1_b, g.bn1_m, g.bn1_v, g.bn1_g, g.bn1_b, 3, s);
  gemm(ws + GF1, 4 * C_, g.conv2_w, 4 * C_, ws + GLOB, C_, B_, C_, 4 * C_,
       g.conv2_b, g.bn2_m, g.bn2_v, g.bn2_g, g.bn2_b, 3, s);

  // ---- GEM (attention collapses: keys/values constant => uniform softmax) ----
  gemm(ws + GLOB, C_, e.fam_in_w + 1024 * C_, C_, ws + T0, C_, B_, C_, C_,
       e.fam_in_b + 1024, 0, 0, 0, 0, 0, s);                       // v proj (fam)
  gemm(ws + T0, C_, e.fam_out_w, C_, ws + T1, C_, B_, C_, C_,
       e.fam_out_b, 0, 0, 0, 0, 0, s);                             // feat (per batch)
  gemm(ws + T1, C_, e.arm_in_w + 1024 * C_, C_, ws + T0, C_, B_, C_, C_,
       e.arm_in_b + 1024, 0, 0, 0, 0, 0, s);                       // v proj (arm)
  gemm(ws + T0, C_, e.arm_out_w, C_, ws + T2, C_, B_, C_, C_,
       e.arm_out_b, 0, 0, 0, 0, 0, s);                             // fa -> T2
  gemm(ws + T2, C_, e.ff1_w, C_, ws + H1, 4 * C_, B_, 4 * C_, C_,
       e.ff1_b, 0, 0, 0, 0, 1, s);                                 // FFN up + ReLU
  gemm(ws + H1, 4 * C_, e.ff2_w, 4 * C_, ws + T0, C_, B_, C_, 4 * C_,
       e.ff2_b, 0, 0, 0, 0, 0, s);                                 // FFN down
  k_add_ln<<<B_, 256, 0, s>>>(ws + T2, ws + T0, e.ln_g, e.ln_b, ws + EV); // edge vec

  // ---- GNN / PNAConv (messages decompose: m = P1[v] + P2[u] + P3[b]) ----
  gemm(ws + EV, C_, n.edge_w, C_, ws + T0, C_, B_, C_, C_,
       n.edge_b, 0, 0, 0, 0, 0, s);                                // eb -> T0
  gemm(xin, C_, n.pre_w, 3 * C_, ws + P1O, C_, B_ * N_, C_, C_,
       nullptr, 0, 0, 0, 0, 0, s);                                 // P1 = x @ W1^T
  gemm(xin, C_, n.pre_w + C_, 3 * C_, ws + P2O, C_, B_ * N_, C_, C_,
       nullptr, 0, 0, 0, 0, 0, s);                                 // P2 = x @ W2^T
  gemm(ws + T0, C_, n.pre_w + 2 * C_, 3 * C_, ws + T1, C_, B_, C_, C_,
       n.pre_b, 0, 0, 0, 0, 0, s);                                 // P3 = eb @ W3^T + b
  k_pna_stats<<<(B_ * C_) / 256, 256, 0, s>>>(ws + P2O, ws + RM, ws + RMIN,
                                              ws + RMAX, ws + RSTD);
  k_assemble<<<(B_ * N_ * 5 * C_) / 256, 256, 0, s>>>(xin, ws + P1O, ws + T1,
      ws + RM, ws + RMIN, ws + RMAX, ws + RSTD, ws + AGG);
  gemm(ws + AGG, 5 * C_, n.post_w, 5 * C_, ws + O1, C_, B_ * N_, C_, 5 * C_,
       n.post_b, 0, 0, 0, 0, 0, s);
  gemm(ws + O1, C_, n.lin_w, C_, ws + O2, C_, B_ * N_, C_, C_,
       n.lin_b, 0, 0, 0, 0, 0, s);
  k_graphnorm<<<(B_ * C_) / 256, 256, 0, s>>>(ws + O2, xin, extra_res,
      n.gn_a, n.gn_b, n.gn_g, outbuf);
}

extern "C" void kernel_launch(void* const* d_in, const int* in_sizes, int n_in,
                              void* d_out, int out_size, void* d_ws, size_t ws_size,
                              hipStream_t stream)
{
  (void)in_sizes; (void)n_in; (void)out_size; (void)ws_size;
  // jax tree leaf order (sorted dict keys): params{e1,e2,g1,g2,n1,n2}, then x
  GemP e1, e2; GfemP g1, g2; GnnP n1, n2;
  const float** dst[6] = { (const float**)&e1, (const float**)&e2,
                           (const float**)&g1, (const float**)&g2,
                           (const float**)&n1, (const float**)&n2 };
  const int cnt[6] = { 14, 14, 12, 12, 11, 11 };
  int i = 0;
  for (int s = 0; s < 6; ++s)
    for (int k = 0; k < cnt[s]; ++k)
      dst[s][k] = (const float*)d_in[i++];
  const float* x = (const float*)d_in[i++];

  float* ws = (float*)d_ws;
  run_stage(x,        g1, e1, n1, nullptr, ws + FV1,      ws, stream); // fv1
  run_stage(ws + FV1, g2, e2, n2, x,       (float*)d_out, ws, stream); // fv2 + x
}
